// Kattention_73031623901707
// MI455X (gfx1250) — matrix operand order repaired and checked by reading, not compiled, not run on hardware
//
#include <hip/hip_runtime.h>
#include <hip/hip_bf16.h>
#include <math.h>

typedef __bf16 bf16_t;
typedef __attribute__((ext_vector_type(16))) __bf16 v16bf;
typedef __attribute__((ext_vector_type(8)))  __bf16 v8bf;
typedef __attribute__((ext_vector_type(4)))  __bf16 v4bf;
typedef __attribute__((ext_vector_type(8)))  float  v8f;

#define B_  2
#define T_  2048
#define C_  768
#define H_  12
#define HD_ 64

// log(2048) / sqrt(64)
#define LSCALE 0.9530773732699248f

// ---- fragment loaders: two 16-byte vector loads, no half-register moves
__device__ __forceinline__ v16bf ld_gap16(const bf16_t* p) {
  // A-matrix 16x32 bf16 lane pattern: 8 elems, skip 8, 8 elems
  v8bf lo = *(const v8bf*)p;
  v8bf hi = *(const v8bf*)(p + 16);
  return __builtin_shufflevector(lo, hi, 0, 1, 2, 3, 4, 5, 6, 7,
                                 8, 9, 10, 11, 12, 13, 14, 15);
}
__device__ __forceinline__ v16bf ld_seq16(const bf16_t* p) {
  v8bf lo = *(const v8bf*)p;
  v8bf hi = *(const v8bf*)(p + 8);
  return __builtin_shufflevector(lo, hi, 0, 1, 2, 3, 4, 5, 6, 7,
                                 8, 9, 10, 11, 12, 13, 14, 15);
}

// branchless top-4 insert (pure cndmask chains, no exec-mask branches)
__device__ __forceinline__ void ins_top4(float sv, int key, bool ok,
                                         float s4[4], int k4[4]) {
  int mi = 0; float mv = s4[0];
  if (s4[1] < mv) { mv = s4[1]; mi = 1; }
  if (s4[2] < mv) { mv = s4[2]; mi = 2; }
  if (s4[3] < mv) { mv = s4[3]; mi = 3; }
  const bool take = ok && (sv > mv);
#pragma unroll
  for (int e = 0; e < 4; ++e) {
    const bool sel = take && (mi == e);
    s4[e] = sel ? sv  : s4[e];
    k4[e] = sel ? key : k4[e];
  }
}

// ---------------------------------------------------------------------------
// GEMM: C[M,N] = A[M,K] * B[K,N]  (f32 in, bf16 WMMA core, f32 accum)
// BF16OUT: bf16 store with optional per-(col mod 64) scale (fused Q scaling).
// Tile 128x128, K-step 32; 256 threads = 8 waves; wave owns 32x64.
// Register double-buffered global->LDS staging.
// ---------------------------------------------------------------------------
#define TM 128
#define TN 128
#define TK 32
#define KPAD 8   // row stride 40 bf16 = 80 B (16B-aligned for b128 frag loads)

template <bool BF16OUT>
__global__ __launch_bounds__(256) void gemm_wmma(
    const float* __restrict__ A, const float* __restrict__ Bm,
    float* __restrict__ Cf, bf16_t* __restrict__ Cb,
    const float* __restrict__ scl, float premul,
    int M, int N, int K)
{
  __shared__ bf16_t As [TM][TK + KPAD];   // A tile, row-major (K contiguous)
  __shared__ bf16_t Bst[TN][TK + KPAD];   // B tile, TRANSPOSED (K contiguous)

  const int tid  = threadIdx.x;
  const int lane = tid & 31;
  const int wave = tid >> 5;
  const int wm   = wave >> 1;   // 0..3 : 32-row slab
  const int wn   = wave & 1;    // 0..1 : 64-col slab

  const int m0 = blockIdx.y * TM;
  const int n0 = blockIdx.x * TN;

  v8f acc[2][4];
#pragma unroll
  for (int s = 0; s < 2; ++s)
#pragma unroll
    for (int j = 0; j < 4; ++j) { v8f z = {}; acc[s][j] = z; }

  const int arow = lane & 15;
  const int ab   = (lane >> 4) << 3;   // 0 or 8  (A-frag K sub-base)
  const int bcol = lane & 15;
  const int bkb  = (lane >> 4) << 3;   // 0 or 8  (B-frag K sub-base)

  // per-thread staging coordinates (idx = i*1024 + tid*4)
  const int aR[4] = {(0 * 1024 + tid * 4) >> 5, (1 * 1024 + tid * 4) >> 5,
                     (2 * 1024 + tid * 4) >> 5, (3 * 1024 + tid * 4) >> 5};
  const int aC   = (tid * 4) & 31;
  const int bR[4] = {(0 * 1024 + tid * 4) >> 7, (1 * 1024 + tid * 4) >> 7,
                     (2 * 1024 + tid * 4) >> 7, (3 * 1024 + tid * 4) >> 7};
  const int bC   = (tid * 4) & 127;

  float4 av[4], bv[4], av2[4], bv2[4];
#pragma unroll
  for (int i = 0; i < 4; ++i) {
    av[i] = *(const float4*)&A [(size_t)(m0 + aR[i]) * K + aC];
    bv[i] = *(const float4*)&Bm[(size_t)(bR[i])      * N + n0 + bC];
  }

  for (int k0 = 0; k0 < K; k0 += TK) {
    // ---- convert + store current tile to LDS
#pragma unroll
    for (int i = 0; i < 4; ++i) {
      v4bf t = {(bf16_t)av[i].x, (bf16_t)av[i].y, (bf16_t)av[i].z, (bf16_t)av[i].w};
      *(v4bf*)&As[aR[i]][aC] = t;                  // ds_store_b64
      Bst[bC + 0][bR[i]] = (bf16_t)bv[i].x;        // transposed scatter
      Bst[bC + 1][bR[i]] = (bf16_t)bv[i].y;
      Bst[bC + 2][bR[i]] = (bf16_t)bv[i].z;
      Bst[bC + 3][bR[i]] = (bf16_t)bv[i].w;
    }
    __syncthreads();

    // ---- issue next tile's global loads (overlap with frags + WMMA)
    const int kn = (k0 + TK < K) ? (k0 + TK) : k0;  // clamped harmless reload
#pragma unroll
    for (int i = 0; i < 4; ++i) {
      av2[i] = *(const float4*)&A [(size_t)(m0 + aR[i]) * K + kn + aC];
      bv2[i] = *(const float4*)&Bm[(size_t)(kn + bR[i]) * N + n0 + bC];
    }
    if (k0 + 2 * TK < K)  // L2 warmup two tiles ahead (global_prefetch_b8)
      __builtin_prefetch(&A[(size_t)(m0 + (tid >> 1)) * K + (k0 + 2 * TK)], 0, 1);

    // ---- fragments: pure b128 LDS loads
    v16bf afrag[2], bfrag[4];
#pragma unroll
    for (int s = 0; s < 2; ++s)
      afrag[s] = ld_gap16(&As[wm * 32 + s * 16 + arow][ab]);
#pragma unroll
    for (int j = 0; j < 4; ++j)
      bfrag[j] = ld_gap16(&Bst[wn * 64 + j * 16 + bcol][bkb]);

#pragma unroll
    for (int s = 0; s < 2; ++s)
#pragma unroll
      for (int j = 0; j < 4; ++j)
        acc[s][j] = __builtin_amdgcn_wmma_f32_16x16x32_bf16(
            false, afrag[s], false, bfrag[j], (short)0, acc[s][j], false, false);

    __syncthreads();
#pragma unroll
    for (int i = 0; i < 4; ++i) { av[i] = av2[i]; bv[i] = bv2[i]; }
  }

  // ---- epilogue (C layout: lanes 0-15 -> M=g, lanes 16-31 -> M=g+8)
  const int cn = lane & 15;
  const int cm = (lane >> 4) << 3;
#pragma unroll
  for (int s = 0; s < 2; ++s)
#pragma unroll
    for (int j = 0; j < 4; ++j) {
      const int n = n0 + wn * 64 + j * 16 + cn;
      float f = 1.0f;
      if (BF16OUT) f = scl ? premul * scl[n & 63] : 1.0f;
#pragma unroll
      for (int g = 0; g < 8; ++g) {
        const size_t off = (size_t)(m0 + wm * 32 + s * 16 + cm + g) * N + n;
        if (BF16OUT) Cb[off] = (bf16_t)(acc[s][j][g] * f);
        else         Cf[off] = acc[s][j][g];
      }
    }
}

// ---------------------------------------------------------------------------
// Attention: per (b,h), 32 query rows per wave. Q already scaled+bf16, K bf16.
// QK^T via WMMA (K-tile fragments software-pipelined); causal + top-4 + tanh
// per row in VALU (vector LDS reads, branchless insert); y = 4-sparse att@V.
// ---------------------------------------------------------------------------
__global__ __launch_bounds__(32) void attn_topk_wmma(
    const bf16_t* __restrict__ qp, const bf16_t* __restrict__ kp,
    const float* __restrict__ vp, float* __restrict__ yp)
{
  __shared__ __align__(16) float sc[2][16][20];  // 80B row stride, 16B aligned

  const int lane  = threadIdx.x;              // 0..31
  const int qbase = blockIdx.x * 32;
  const int bh    = blockIdx.y;
  const int b     = bh / H_;
  const int h     = bh % H_;
  const size_t rowbase = (size_t)b * T_;      // row offset into [B*T, C] buffers
  const int hoff  = h * HD_;

  // ---- Q fragments: straight b128 loads (scaling fused in projection GEMM)
  v16bf aq[2][2];
  {
    const int am = lane & 15;
    const int ab = (lane >> 4) << 3;
#pragma unroll
    for (int s = 0; s < 2; ++s) {
      const bf16_t* qr = qp + (rowbase + qbase + s * 16 + am) * C_ + hoff;
#pragma unroll
      for (int kc = 0; kc < 2; ++kc)
        aq[s][kc] = ld_gap16(qr + kc * 32 + ab);
    }
  }

  // ---- per-lane row state (lane owns query row qbase+lane)
  const int qrow = qbase + lane;
  const int sub  = lane >> 4;
  const int mm   = lane & 15;
  float s4[4] = {0.0f, 0.0f, 0.0f, 0.0f};    // masked entries are 0 in ref
  int   k4[4] = {-1, -1, -1, -1};

  const int bn = lane & 15;
  const int bd = (lane >> 4) << 3;
  const int ntile = (qbase >> 4) + 2;        // key tiles covering keys <= qbase+31

  // prologue: load K^T fragments for tile 0
  v16bf bk0, bk1;
  {
    const bf16_t* kr = kp + (rowbase + bn) * C_ + hoff;
    bk0 = ld_gap16(kr + bd);
    bk1 = ld_gap16(kr + 32 + bd);
  }

  for (int t = 0; t < ntile; ++t) {
    const int kb0 = t * 16;

    // scores: two 16x16 tiles, K=64 contraction = 2 chained WMMAs each
#pragma unroll
    for (int s = 0; s < 2; ++s) {
      v8f c = {};
      c = __builtin_amdgcn_wmma_f32_16x16x32_bf16(false, aq[s][0], false, bk0,
                                                  (short)0, c, false, false);
      c = __builtin_amdgcn_wmma_f32_16x16x32_bf16(false, aq[s][1], false, bk1,
                                                  (short)0, c, false, false);
      const int cm = (lane >> 4) << 3;
      const int cn = lane & 15;
#pragma unroll
      for (int g = 0; g < 8; ++g) sc[s][cm + g][cn] = c[g];
    }

    // pipeline: issue next K-tile fragment loads (fly during the scan)
    {
      int krow = kb0 + 16 + bn;
      krow = (krow < T_) ? krow : (T_ - 1);  // clamped harmless load on last iter
      const bf16_t* kr = kp + (rowbase + krow) * C_ + hoff;
      bk0 = ld_gap16(kr + bd);
      bk1 = ld_gap16(kr + 32 + bd);
    }

    // in-wave LDS visibility (gfx1250 split-counter wait)
    asm volatile("s_wait_dscnt 0" ::: "memory");

    // each lane scans its own row: 4x ds_load_b128, 16 branchless inserts
    const float* srow = &sc[sub][mm][0];
    float4 r0 = *(const float4*)(srow + 0);
    float4 r1 = *(const float4*)(srow + 4);
    float4 r2 = *(const float4*)(srow + 8);
    float4 r3 = *(const float4*)(srow + 12);
    ins_top4(r0.x, kb0 + 0,  kb0 + 0  <= qrow, s4, k4);
    ins_top4(r0.y, kb0 + 1,  kb0 + 1  <= qrow, s4, k4);
    ins_top4(r0.z, kb0 + 2,  kb0 + 2  <= qrow, s4, k4);
    ins_top4(r0.w, kb0 + 3,  kb0 + 3  <= qrow, s4, k4);
    ins_top4(r1.x, kb0 + 4,  kb0 + 4  <= qrow, s4, k4);
    ins_top4(r1.y, kb0 + 5,  kb0 + 5  <= qrow, s4, k4);
    ins_top4(r1.z, kb0 + 6,  kb0 + 6  <= qrow, s4, k4);
    ins_top4(r1.w, kb0 + 7,  kb0 + 7  <= qrow, s4, k4);
    ins_top4(r2.x, kb0 + 8,  kb0 + 8  <= qrow, s4, k4);
    ins_top4(r2.y, kb0 + 9,  kb0 + 9  <= qrow, s4, k4);
    ins_top4(r2.z, kb0 + 10, kb0 + 10 <= qrow, s4, k4);
    ins_top4(r2.w, kb0 + 11, kb0 + 11 <= qrow, s4, k4);
    ins_top4(r3.x, kb0 + 12, kb0 + 12 <= qrow, s4, k4);
    ins_top4(r3.y, kb0 + 13, kb0 + 13 <= qrow, s4, k4);
    ins_top4(r3.z, kb0 + 14, kb0 + 14 <= qrow, s4, k4);
    ins_top4(r3.w, kb0 + 15, kb0 + 15 <= qrow, s4, k4);

    // lanes finished reading their row; LDS reuse next iter is in-order per wave
  }

  // ---- y[row] = sum_j tanh(s_j) * V[key_j]  (4-sparse att@V)
  float  w[4];
  size_t vrow[4];
#pragma unroll
  for (int j = 0; j < 4; ++j) {
    w[j] = (k4[j] >= 0) ? tanhf(s4[j]) : 0.0f;
    const int kk = (k4[j] < 0) ? 0 : k4[j];
    vrow[j] = (rowbase + kk) * C_ + hoff;
  }
  float* yr = yp + (rowbase + qrow) * C_ + hoff;
#pragma unroll
  for (int d = 0; d < HD_; d += 4) {
    float4 a0 = *(const float4*)(vp + vrow[0] + d);
    float4 a1 = *(const float4*)(vp + vrow[1] + d);
    float4 a2 = *(const float4*)(vp + vrow[2] + d);
    float4 a3 = *(const float4*)(vp + vrow[3] + d);
    float4 o;
    o.x = w[0] * a0.x + w[1] * a1.x + w[2] * a2.x + w[3] * a3.x;
    o.y = w[0] * a0.y + w[1] * a1.y + w[2] * a2.y + w[3] * a3.y;
    o.z = w[0] * a0.z + w[1] * a1.z + w[2] * a2.z + w[3] * a3.z;
    o.w = w[0] * a0.w + w[1] * a1.w + w[2] * a2.w + w[3] * a3.w;
    *(float4*)(yr + d) = o;
  }
}

// ---------------------------------------------------------------------------
extern "C" void kernel_launch(void* const* d_in, const int* in_sizes, int n_in,
                              void* d_out, int out_size, void* d_ws, size_t ws_size,
                              hipStream_t stream) {
  (void)in_sizes; (void)n_in; (void)out_size; (void)ws_size;
  const float* x  = (const float*)d_in[0];
  const float* Wq = (const float*)d_in[1];
  const float* Wk = (const float*)d_in[2];
  const float* Wv = (const float*)d_in[3];
  const float* Wp = (const float*)d_in[4];
  const float* qm = (const float*)d_in[5];
  float* out = (float*)d_out;

  const size_t SZ = (size_t)B_ * T_ * C_;
  bf16_t* qbf  = (bf16_t*)d_ws;           // Q (scaled), bf16
  bf16_t* kbf  = qbf + SZ;                // K, bf16
  float*  vbuf = (float*)(kbf + SZ);      // V, f32
  float*  ybuf = vbuf + SZ;               // attention out, f32

  const int M = B_ * T_;          // 4096
  dim3 gg(C_ / TN, M / TM);       // 6 x 32
  dim3 gb(256);

  // Q projection with fused log(T)*qm/sqrt(HD) scaling, bf16 out
  gemm_wmma<true ><<<gg, gb, 0, stream>>>(x, Wq, nullptr, qbf, qm, LSCALE,
                                          M, C_, C_);
  // K projection, bf16 out
  gemm_wmma<true ><<<gg, gb, 0, stream>>>(x, Wk, nullptr, kbf, nullptr, 1.0f,
                                          M, C_, C_);
  // V projection, f32 out
  gemm_wmma<false><<<gg, gb, 0, stream>>>(x, Wv, vbuf, nullptr, nullptr, 1.0f,
                                          M, C_, C_);

  dim3 ag(T_ / 32, B_ * H_);      // 64 x 24 waves
  attn_topk_wmma<<<ag, dim3(32), 0, stream>>>(qbf, kbf, vbuf, ybuf);

  // output projection, f32 out
  gemm_wmma<false><<<gg, gb, 0, stream>>>(ybuf, Wp, out, nullptr, nullptr, 1.0f,
                                          M, C_, C_);
}
